// MoIR_7799660609919
// MI455X (gfx1250) — compile-verified, matrix operand order checked by reading, not run on hardware
//
#include <hip/hip_runtime.h>
#include <hip/hip_bf16.h>
#include <cstdint>
#include <cstddef>

// Problem constants (match reference)
#define BZ    8
#define SQ    2048
#define DD    1024
#define QDIM  64
#define KLOW  102
#define EPSF  1e-6f
#define NITER 6

typedef __attribute__((ext_vector_type(16))) __bf16         v16bf;
typedef __attribute__((ext_vector_type(8)))  float          v8f;
typedef __attribute__((ext_vector_type(8)))  unsigned short v8us;
typedef __attribute__((ext_vector_type(16))) unsigned short v16us;

__device__ __forceinline__ unsigned short f2bf_rne(float f) {
  unsigned u = __float_as_uint(f);
  unsigned r = (u + 0x7fffu + ((u >> 16) & 1u)) >> 16;
  return (unsigned short)r;
}

// Low 32 bits of a generic pointer into LDS == LDS byte offset (ISA §10.2:
// LDS aperture addresses truncate to addr[31:0]).
__device__ __forceinline__ unsigned lds_off32(const void* p) {
  return (unsigned)(unsigned long long)p;
}

// ---------------------------------------------------------------------------
// Utility: zero a float buffer
// ---------------------------------------------------------------------------
__global__ void zero_f32(float* __restrict__ p, int n) {
  int i = blockIdx.x * 256 + threadIdx.x;
  if (i < n) p[i] = 0.f;
}

// ---------------------------------------------------------------------------
// n[b] = max(sum_s mask[b,s], 1)
// ---------------------------------------------------------------------------
__global__ void count_n_kernel(const unsigned char* __restrict__ mask,
                               float* __restrict__ nbuf) {
  __shared__ float red[256];
  int b = blockIdx.x, t = threadIdx.x;
  float c = 0.f;
  for (int s = t; s < SQ; s += 256) c += mask[(size_t)b * SQ + s] ? 1.f : 0.f;
  red[t] = c; __syncthreads();
  for (int o = 128; o > 0; o >>= 1) { if (t < o) red[t] += red[t + o]; __syncthreads(); }
  if (t == 0) nbuf[b] = fmaxf(red[0], 1.f);
}

// ---------------------------------------------------------------------------
// Masked sums over S (chunked, atomic accumulate): sumBuf/sqBuf [side][b][d]
// ---------------------------------------------------------------------------
__global__ __launch_bounds__(256) void stats_partial(
    const float* __restrict__ A, const float* __restrict__ Bv,
    const unsigned char* __restrict__ mask,
    float* __restrict__ sumBuf, float* __restrict__ sqBuf) {
  int b = blockIdx.x;
  int d = blockIdx.y * 256 + threadIdx.x;
  int s0 = blockIdx.z * (SQ / 8);
  size_t base = ((size_t)b * SQ) * DD + d;
  float sa = 0.f, qa = 0.f, sb = 0.f, qb = 0.f;
  for (int s = s0; s < s0 + SQ / 8; ++s) {
    if (s + 32 < s0 + SQ / 8) {
      __builtin_prefetch(&A[base + (size_t)(s + 32) * DD], 0, 0);
      __builtin_prefetch(&Bv[base + (size_t)(s + 32) * DD], 0, 0);
    }
    float m = mask[(size_t)b * SQ + s] ? 1.f : 0.f;
    float av = A[base + (size_t)s * DD];
    float bv = Bv[base + (size_t)s * DD];
    sa += av * m; qa += av * av * m;
    sb += bv * m; qb += bv * bv * m;
  }
  atomicAdd(&sumBuf[(size_t)b * DD + d], sa);
  atomicAdd(&sumBuf[(size_t)(BZ + b) * DD + d], sb);
  atomicAdd(&sqBuf[(size_t)b * DD + d], qa);
  atomicAdd(&sqBuf[(size_t)(BZ + b) * DD + d], qb);
}

// sums -> mean (in sumBuf), rms (in sqBuf)
__global__ void finalize_stats(const float* __restrict__ nbuf,
                               float* __restrict__ sumBuf, float* __restrict__ sqBuf) {
  int b = blockIdx.x;
  int d = blockIdx.y * 256 + threadIdx.x;
  float n = nbuf[b];
  for (int side = 0; side < 2; ++side) {
    size_t i = (size_t)(side * BZ + b) * DD + d;
    sumBuf[i] = sumBuf[i] / n;
    sqBuf[i]  = fmaxf(sqrtf(sqBuf[i] / n), EPSF);
  }
}

// ---------------------------------------------------------------------------
// Xc[side][b][s][d] = bf16((x - mean) * mask)
// ---------------------------------------------------------------------------
__global__ void center_kernel(const float* __restrict__ A, const float* __restrict__ Bv,
                              const unsigned char* __restrict__ mask,
                              const float* __restrict__ meanBuf,
                              unsigned short* __restrict__ Xc) {
  size_t idx = (size_t)blockIdx.x * 256 + threadIdx.x;   // < BZ*SQ*DD
  int d = (int)(idx % DD);
  size_t bs = idx / DD;
  int s = (int)(bs % SQ);
  int b = (int)(bs / SQ);
  float m = mask[(size_t)b * SQ + s] ? 1.f : 0.f;
  Xc[idx] = f2bf_rne((A[idx]  - meanBuf[(size_t)b * DD + d]) * m);
  Xc[(size_t)BZ * SQ * DD + idx] =
      f2bf_rne((Bv[idx] - meanBuf[(size_t)(BZ + b) * DD + d]) * m);
}

// ---------------------------------------------------------------------------
// Gram GEMM: G = Xc^T * Xc  (1024x1024, K=2048, bf16 in / fp32 acc)
// grid (36 upper-tri 128x128 tiles, B, 2 sides), block 256 = 8 waves.
// Wave tile = 32(M) x 64(N) = 2x4 WMMA 16x16 accumulators.
// Double-buffered LDS: compute buf while staging buf^1 (one barrier/step).
// ---------------------------------------------------------------------------
#define GT   128   // workgroup tile edge
#define GKT  32    // K step
#define LDP  40    // padded LDS row length (ushorts); 80B rows keep 16B align

// stage 32(K) x 128(ch) tiles, stored transposed [ch][K] into dstA/dstB
#define STAGE_TILE(KKBASE, DSTA, DSTB)                                        \
  do {                                                                        \
    _Pragma("unroll")                                                         \
    for (int it_ = 0; it_ < 4; ++it_) {                                       \
      int idx_ = it_ * 256 + tid;                                             \
      int k_  = idx_ >> 5;                                                    \
      int c4_ = (idx_ & 31) * 4;                                              \
      unsigned long long va_ =                                                \
          *(const unsigned long long*)(X + (size_t)((KKBASE) + k_) * DD + gi + c4_); \
      (DSTA)[(c4_ + 0) * LDP + k_] = (unsigned short)(va_);                   \
      (DSTA)[(c4_ + 1) * LDP + k_] = (unsigned short)(va_ >> 16);             \
      (DSTA)[(c4_ + 2) * LDP + k_] = (unsigned short)(va_ >> 32);             \
      (DSTA)[(c4_ + 3) * LDP + k_] = (unsigned short)(va_ >> 48);             \
      unsigned long long vb_ =                                                \
          *(const unsigned long long*)(X + (size_t)((KKBASE) + k_) * DD + gj + c4_); \
      (DSTB)[(c4_ + 0) * LDP + k_] = (unsigned short)(vb_);                   \
      (DSTB)[(c4_ + 1) * LDP + k_] = (unsigned short)(vb_ >> 16);             \
      (DSTB)[(c4_ + 2) * LDP + k_] = (unsigned short)(vb_ >> 32);             \
      (DSTB)[(c4_ + 3) * LDP + k_] = (unsigned short)(vb_ >> 48);             \
    }                                                                         \
  } while (0)

__global__ __launch_bounds__(256) void gram_kernel(
    const unsigned short* __restrict__ Xc, float* __restrict__ G) {
  int side = blockIdx.z, b = blockIdx.y;
  // map linear tile id -> (bi, bj) with bi <= bj over an 8x8 grid
  int rem = blockIdx.x, bi = 0;
  while (rem >= (8 - bi)) { rem -= (8 - bi); ++bi; }
  int bj = bi + rem;
  int gi = bi * GT, gj = bj * GT;

  const unsigned short* X = Xc + ((size_t)(side * BZ + b) * SQ) * DD;
  float* Gp = G + ((size_t)(side * BZ + b) * DD) * DD;

  __shared__ __attribute__((aligned(16))) unsigned short tA[2][GT * LDP];
  __shared__ __attribute__((aligned(16))) unsigned short tB[2][GT * LDP];

  int tid = threadIdx.x;
  int lane = tid & 31;
  int wave = tid >> 5;
  int m_off = (wave >> 1) * 32;   // 0,32,64,96
  int n_off = (wave & 1) * 64;    // 0,64
  int cidx  = lane & 15;
  int bk    = (lane >> 4) * 8;    // K-half offset per ISA A/B 16-bit layout

  v8f zero8 = {0.f, 0.f, 0.f, 0.f, 0.f, 0.f, 0.f, 0.f};
  v8f acc[2][4];
#pragma unroll
  for (int m = 0; m < 2; ++m)
#pragma unroll
    for (int n = 0; n < 4; ++n) acc[m][n] = zero8;

  // prologue: stage step 0 into buffer 0
  STAGE_TILE(0, tA[0], tB[0]);
  __syncthreads();

  const int nsteps = SQ / GKT;   // 64
  for (int it = 0; it < nsteps; ++it) {
    int buf = it & 1;
    const unsigned short* cA = tA[buf];
    const unsigned short* cB = tB[buf];

    // Build fragments: per lane 16 bf16 K-values of one channel row:
    // elems 0..7 = K bk..bk+7, elems 8..15 = K bk+16..bk+23 (contiguous runs)
    v16bf aF[2], bF[4];
#pragma unroll
    for (int mt = 0; mt < 2; ++mt) {
      int off = (m_off + mt * 16 + cidx) * LDP + bk;
      v8us lo = *(const v8us*)&cA[off];
      v8us hi = *(const v8us*)&cA[off + 16];
      v16us u;
#pragma unroll
      for (int i = 0; i < 8; ++i) { u[i] = lo[i]; u[i + 8] = hi[i]; }
      aF[mt] = __builtin_bit_cast(v16bf, u);
    }
#pragma unroll
    for (int nt = 0; nt < 4; ++nt) {
      int off = (n_off + nt * 16 + cidx) * LDP + bk;
      v8us lo = *(const v8us*)&cB[off];
      v8us hi = *(const v8us*)&cB[off + 16];
      v16us u;
#pragma unroll
      for (int i = 0; i < 8; ++i) { u[i] = lo[i]; u[i + 8] = hi[i]; }
      bF[nt] = __builtin_bit_cast(v16bf, u);
    }

#pragma unroll
    for (int mt = 0; mt < 2; ++mt)
#pragma unroll
      for (int nt = 0; nt < 4; ++nt)
        acc[mt][nt] = __builtin_amdgcn_wmma_f32_16x16x32_bf16(
            false, aF[mt], false, bF[nt], (short)0, acc[mt][nt], false, false);

    // stage next step into the other buffer while WMMAs drain
    if (it + 1 < nsteps) {
      int kkn = (it + 1) * GKT;
      STAGE_TILE(kkn, tA[buf ^ 1], tB[buf ^ 1]);
    }
    __syncthreads();
  }

  // C/D layout: VGPR r, lanes 0-15: (M=r, N=lane); lanes 16-31: (M=r+8, N=lane-16)
  int m_add = (lane >= 16) ? 8 : 0;
#pragma unroll
  for (int mt = 0; mt < 2; ++mt)
#pragma unroll
    for (int nt = 0; nt < 4; ++nt)
#pragma unroll
      for (int r = 0; r < 8; ++r) {
        int m = gi + m_off + mt * 16 + r + m_add;
        int n = gj + n_off + nt * 16 + cidx;
        float v = acc[mt][nt][r];
        Gp[(size_t)m * DD + n] = v;   // tile
        Gp[(size_t)n * DD + m] = v;   // symmetric mirror
      }
}

// ---------------------------------------------------------------------------
// Deterministic pseudo-random init of Q0
// ---------------------------------------------------------------------------
__global__ void init_q(float* __restrict__ Q) {
  size_t idx = (size_t)blockIdx.x * 256 + threadIdx.x;   // 2*BZ*DD*QDIM
  unsigned h = (unsigned)idx * 2654435761u;
  h ^= h >> 16; h *= 2246822519u; h ^= h >> 13;
  Q[idx] = ((float)(h & 0xffff) / 65536.0f) - 0.5f;
}

// ---------------------------------------------------------------------------
// Y = G * Q   (D x QDIM, K = D, fp32)
// grid (D/4, B, 2), block 256 -> each block: 4 rows x 64 cols.
// Q K-chunks staged into LDS with GLOBAL_LOAD_ASYNC_TO_LDS_B128 (ASYNCcnt).
// ---------------------------------------------------------------------------
#define GQ_CH 128   // K chunk: 128 x 64 f32 = 32 KB, contiguous in global

__global__ __launch_bounds__(256) void gemm_gq(
    const float* __restrict__ G, const float* __restrict__ Qin,
    float* __restrict__ Yout) {
  int sb = blockIdx.z * BZ + blockIdx.y;
  int t = threadIdx.x;
  int row = blockIdx.x * 4 + (t >> 6);
  int j = t & 63;
  const float* Gp = G + (size_t)sb * DD * DD + (size_t)row * DD;
  const float* Qp = Qin + (size_t)sb * DD * QDIM;

  __shared__ __attribute__((aligned(16))) float Qs[GQ_CH * QDIM];  // 32 KB

  float acc = 0.f;
  for (int k0 = 0; k0 < DD; k0 += GQ_CH) {
    __syncthreads();   // previous chunk's compute done before overwrite
    // async copy 32 KB (2048 x 16 B), 8 transfers per thread
#pragma unroll
    for (int i = 0; i < 8; ++i) {
      int lin = i * 256 + t;                         // 0..2047
      unsigned lds = lds_off32(&Qs[lin * 4]);
      unsigned long long ga =
          (unsigned long long)(const void*)(Qp + (size_t)k0 * QDIM + (size_t)lin * 4);
      asm volatile("global_load_async_to_lds_b128 %0, %1, off"
                   :: "v"(lds), "v"(ga) : "memory");
    }
    asm volatile("s_wait_asynccnt 0x0" ::: "memory");
    __syncthreads();
    if (j == 0) __builtin_prefetch(&Gp[k0 + GQ_CH], 0, 0);
#pragma unroll 4
    for (int k = 0; k < GQ_CH; ++k)
      acc += Gp[k0 + k] * Qs[k * QDIM + j];
  }
  Yout[(size_t)sb * DD * QDIM + (size_t)row * QDIM + j] = acc;
}

// ---------------------------------------------------------------------------
// C += A^T * B over a K-chunk (A,B: D x QDIM; C: QDIM x QDIM), atomic accumulate
// grid (8 K-chunks, B, 2), block 256
// ---------------------------------------------------------------------------
__global__ __launch_bounds__(256) void atAB_partial(
    const float* __restrict__ Aa, const float* __restrict__ Bb,
    float* __restrict__ C) {
  int sb = blockIdx.z * BZ + blockIdx.y;
  const float* Ap = Aa + (size_t)sb * DD * QDIM;
  const float* Bp = Bb + (size_t)sb * DD * QDIM;
  float* Cp = C + (size_t)sb * QDIM * QDIM;
  int k0 = blockIdx.x * (DD / 8);
#pragma unroll 1
  for (int w = 0; w < 16; ++w) {
    int e = w * 256 + threadIdx.x;
    int i = e >> 6, j = e & 63;
    float acc = 0.f;
    for (int k = k0; k < k0 + DD / 8; ++k)
      acc += Ap[(size_t)k * QDIM + i] * Bp[(size_t)k * QDIM + j];
    atomicAdd(&Cp[e], acc);
  }
}

// ---------------------------------------------------------------------------
// In-place Cholesky of 64x64 (lower factor kept). grid (1,B,2), block 64.
// ---------------------------------------------------------------------------
__global__ __launch_bounds__(64) void chol_kernel(float* __restrict__ C) {
  int sb = blockIdx.z * BZ + blockIdx.y;
  float* Cp = C + (size_t)sb * QDIM * QDIM;
  __shared__ float L[QDIM * QDIM];
  int t = threadIdx.x;
  for (int i = 0; i < QDIM; ++i) L[i * QDIM + t] = Cp[i * QDIM + t];
  __syncthreads();
  for (int k = 0; k < QDIM; ++k) {
    float dk = sqrtf(fmaxf(L[k * QDIM + k], 1e-30f));
    float lik = 0.f;
    if (t == k) lik = dk;
    else if (t > k) lik = L[t * QDIM + k] / dk;
    __syncthreads();
    if (t >= k) L[t * QDIM + k] = lik;
    __syncthreads();
    if (t > k) {
      float ltk = L[t * QDIM + k];
      for (int j = k + 1; j <= t; ++j) L[t * QDIM + j] -= ltk * L[j * QDIM + k];
    }
    __syncthreads();
  }
  for (int i = 0; i < QDIM; ++i) Cp[i * QDIM + t] = L[i * QDIM + t];
}

// ---------------------------------------------------------------------------
// Row-parallel forward substitution: Qout[d,:] solves L x = Y[d,:]
// grid (D/128, B, 2), block 128
// ---------------------------------------------------------------------------
__global__ __launch_bounds__(128) void trisolve_kernel(
    const float* __restrict__ Y, const float* __restrict__ C,
    float* __restrict__ Qout) {
  int sb = blockIdx.z * BZ + blockIdx.y;
  const float* Yp = Y + (size_t)sb * DD * QDIM;
  const float* Lg = C + (size_t)sb * QDIM * QDIM;
  float* Qp = Qout + (size_t)sb * DD * QDIM;
  __shared__ float L[QDIM * QDIM];      // 16 KB
  __shared__ float X[QDIM * 128];       // 32 KB, [i][t]
  int t = threadIdx.x;
  for (int i = t; i < QDIM * QDIM; i += 128) L[i] = Lg[i];
  __syncthreads();
  int d = blockIdx.x * 128 + t;
  for (int i = 0; i < QDIM; ++i) {
    float xi = Yp[(size_t)d * QDIM + i];
    for (int j = 0; j < i; ++j) xi -= L[i * QDIM + j] * X[j * 128 + t];
    X[i * 128 + t] = xi / L[i * QDIM + i];
  }
  for (int i = 0; i < QDIM; ++i) Qp[(size_t)d * QDIM + i] = X[i * 128 + t];
}

// ---------------------------------------------------------------------------
// scores[d] = q_d^T * T * q_d   (q_d = row d of Q)
// grid (D/128, B, 2), block 128
// ---------------------------------------------------------------------------
__global__ __launch_bounds__(128) void scores_kernel(
    const float* __restrict__ Q, const float* __restrict__ T,
    float* __restrict__ scores) {
  int sb = blockIdx.z * BZ + blockIdx.y;
  const float* Qp = Q + (size_t)sb * DD * QDIM;
  const float* Tp = T + (size_t)sb * QDIM * QDIM;
  __shared__ float Ts[QDIM * QDIM];     // 16 KB
  __shared__ float qsh[QDIM * 128];     // 32 KB
  int t = threadIdx.x;
  for (int i = t; i < QDIM * QDIM; i += 128) Ts[i] = Tp[i];
  int d = blockIdx.x * 128 + t;
  for (int i = 0; i < QDIM; ++i) qsh[i * 128 + t] = Qp[(size_t)d * QDIM + i];
  __syncthreads();
  float s = 0.f;
  for (int i = 0; i < QDIM; ++i) {
    float qi = qsh[i * 128 + t];
    float p = 0.f;
#pragma unroll
    for (int j = 0; j < QDIM; ++j) p += Ts[i * QDIM + j] * qsh[j * 128 + t];
    s += qi * p;
  }
  scores[(size_t)sb * DD + d] = s;
}

// ---------------------------------------------------------------------------
// Stable bottom-K selection (matches top_k(-scores) tie order)
// grid (1, B, 2), block 256
// ---------------------------------------------------------------------------
__global__ __launch_bounds__(256) void rank_kernel(
    const float* __restrict__ scores, int* __restrict__ bottom) {
  int sb = blockIdx.z * BZ + blockIdx.y;
  const float* sp = scores + (size_t)sb * DD;
  __shared__ float ss[DD];
  int t = threadIdx.x;
  for (int i = t; i < DD; i += 256) ss[i] = sp[i];
  __syncthreads();
  for (int d = t; d < DD; d += 256) {
    float sd = ss[d];
    int r = 0;
    for (int j = 0; j < DD; ++j) {
      float sj = ss[j];
      r += (sj < sd) || (sj == sd && j < d);
    }
    bottom[(size_t)sb * DD + d] = (r < KLOW) ? 1 : 0;
  }
}

// ---------------------------------------------------------------------------
// Final blend for both sides
// ---------------------------------------------------------------------------
__global__ void blend_kernel(const float* __restrict__ A, const float* __restrict__ Bv,
                             const float* __restrict__ alphaLogits,
                             const unsigned char* __restrict__ mask,
                             const float* __restrict__ meanBuf,
                             const float* __restrict__ rmsBuf,
                             const int* __restrict__ bottom,
                             float* __restrict__ out) {
  size_t idx = (size_t)blockIdx.x * 256 + threadIdx.x;
  int d = (int)(idx % DD);
  size_t bs = idx / DD;
  int s = (int)(bs % SQ);
  int b = (int)(bs / SQ);
  float al = 1.f / (1.f + __expf(-alphaLogits[d]));
  bool m = mask[(size_t)b * SQ + s] != 0;
  {
    // side 0: target = a, other = b
    float tv = A[idx];
    float pooled = meanBuf[(size_t)(BZ + b) * DD + d];
    float rms = rmsBuf[(size_t)b * DD + d];
    float sadj = pooled * rms / fmaxf(fabsf(pooled), EPSF);
    bool sel = m && (bottom[(size_t)b * DD + d] != 0);
    out[idx] = sel ? (al * sadj + (1.f - al) * tv) : tv;
  }
  {
    // side 1: target = b, other = a
    float tv = Bv[idx];
    float pooled = meanBuf[(size_t)b * DD + d];
    float rms = rmsBuf[(size_t)(BZ + b) * DD + d];
    float sadj = pooled * rms / fmaxf(fabsf(pooled), EPSF);
    bool sel = m && (bottom[(size_t)(BZ + b) * DD + d] != 0);
    out[(size_t)BZ * SQ * DD + idx] = sel ? (al * sadj + (1.f - al) * tv) : tv;
  }
}

// ---------------------------------------------------------------------------
// Host launcher
// ---------------------------------------------------------------------------
extern "C" void kernel_launch(void* const* d_in, const int* in_sizes, int n_in,
                              void* d_out, int out_size, void* d_ws, size_t ws_size,
                              hipStream_t stream) {
  const float* A  = (const float*)d_in[0];
  const float* Bv = (const float*)d_in[1];
  const float* alphaL = (const float*)d_in[2];
  const unsigned char* mask = (const unsigned char*)d_in[3];
  float* out = (float*)d_out;

  char* ws = (char*)d_ws;
  size_t off = 0;
  auto take = [&](size_t bytes) -> char* {
    char* p = ws + off;
    off = (off + bytes + 255) & ~(size_t)255;
    return p;
  };
  float* meanBuf = (float*)take((size_t)2 * BZ * DD * 4);            // sums -> means
  float* rmsBuf  = (float*)take((size_t)2 * BZ * DD * 4);            // sqsums -> rms
  float* nbuf    = (float*)take((size_t)BZ * 4);
  float* Qa      = (float*)take((size_t)2 * BZ * DD * QDIM * 4);
  float* Qb      = (float*)take((size_t)2 * BZ * DD * QDIM * 4);
  float* Ybuf    = (float*)take((size_t)2 * BZ * DD * QDIM * 4);
  float* Cbuf    = (float*)take((size_t)2 * BZ * QDIM * QDIM * 4);
  float* scores  = (float*)take((size_t)2 * BZ * DD * 4);
  int*   bottom  = (int*)take((size_t)2 * BZ * DD * 4);
  unsigned short* Xc = (unsigned short*)take((size_t)2 * BZ * SQ * DD * 2);
  float* G       = (float*)take((size_t)2 * BZ * DD * DD * 4);

  const int statsN = 2 * BZ * DD;                       // 16384
  const int cN     = 2 * BZ * QDIM * QDIM;              // 65536
  const int elemBlocks = (BZ * SQ * DD) / 256;          // 65536

  // 1) stats
  zero_f32<<<(statsN + 255) / 256, 256, 0, stream>>>(meanBuf, statsN);
  zero_f32<<<(statsN + 255) / 256, 256, 0, stream>>>(rmsBuf, statsN);
  count_n_kernel<<<BZ, 256, 0, stream>>>(mask, nbuf);
  stats_partial<<<dim3(BZ, DD / 256, 8), 256, 0, stream>>>(A, Bv, mask, meanBuf, rmsBuf);
  finalize_stats<<<dim3(BZ, DD / 256), 256, 0, stream>>>(nbuf, meanBuf, rmsBuf);

  // 2) centered bf16 copies
  center_kernel<<<elemBlocks, 256, 0, stream>>>(A, Bv, mask, meanBuf, Xc);

  // 3) Gram matrices via bf16 WMMA (symmetric, upper-triangle tiles)
  gram_kernel<<<dim3(36, BZ, 2), 256, 0, stream>>>(Xc, G);

  // 4) block power iteration with Cholesky-QR
  init_q<<<(2 * BZ * DD * QDIM) / 256, 256, 0, stream>>>(Qa);
  float* Qcur = Qa;
  float* Qnxt = Qb;
  for (int it = 0; it < NITER; ++it) {
    gemm_gq<<<dim3(DD / 4, BZ, 2), 256, 0, stream>>>(G, Qcur, Ybuf);
    zero_f32<<<(cN + 255) / 256, 256, 0, stream>>>(Cbuf, cN);
    atAB_partial<<<dim3(8, BZ, 2), 256, 0, stream>>>(Ybuf, Ybuf, Cbuf);
    chol_kernel<<<dim3(1, BZ, 2), 64, 0, stream>>>(Cbuf);
    trisolve_kernel<<<dim3(DD / 128, BZ, 2), 128, 0, stream>>>(Ybuf, Cbuf, Qnxt);
    float* tmp = Qcur; Qcur = Qnxt; Qnxt = tmp;
  }

  // 5) T = Q^T G Q, scores = diag(Q T Q^T)
  gemm_gq<<<dim3(DD / 4, BZ, 2), 256, 0, stream>>>(G, Qcur, Ybuf);
  zero_f32<<<(cN + 255) / 256, 256, 0, stream>>>(Cbuf, cN);
  atAB_partial<<<dim3(8, BZ, 2), 256, 0, stream>>>(Qcur, Ybuf, Cbuf);
  scores_kernel<<<dim3(DD / 128, BZ, 2), 128, 0, stream>>>(Qcur, Cbuf, scores);

  // 6) bottom-K selection + blend
  rank_kernel<<<dim3(1, BZ, 2), 256, 0, stream>>>(scores, bottom);
  blend_kernel<<<elemBlocks, 256, 0, stream>>>(A, Bv, alphaL, mask, meanBuf, rmsBuf,
                                               bottom, out);
}
